// MuCausalSelfAttention_1795296329869
// MI455X (gfx1250) — compile-verified, matrix operand order checked by reading, not run on hardware
//
#include <hip/hip_runtime.h>
#include <cstdint>

typedef unsigned short u16;
typedef __attribute__((ext_vector_type(16))) __bf16 v16bf;
typedef __attribute__((ext_vector_type(8)))  float  v8f;

#define B_  4
#define T_  2048
#define C_  1024
#define H_  16
#define D_  64
#define C3_ 3072
// softmax scale folded with log2(e): exp(s/64 - m) == exp2(s*SC - m')
#define SC_ (1.44269504f / 64.0f)

__device__ __forceinline__ u16 f2bf(float f) {
    unsigned int u = __float_as_uint(f);
    unsigned int r = u + 0x7FFFu + ((u >> 16) & 1u);
    return (u16)(r >> 16);
}

__device__ __forceinline__ v8f vzero8() {
    v8f v = {0.f,0.f,0.f,0.f,0.f,0.f,0.f,0.f};
    return v;
}

// XOR-lane shuffle via ds_swizzle (group-of-32 mode: and=0x1f, or=0, xor=m<<10).
// Masks 1,2,4,8 reduce within 16-lane halves, matching the WMMA C-tile layout
// (row = vgpr + 8*half, col = lane&15).
#define SWZ_XOR(v, m) (__int_as_float(__builtin_amdgcn_ds_swizzle(__float_as_int(v), (0x1F | ((m) << 10)))))

__device__ __forceinline__ float redmax16(float v) {
    v = fmaxf(v, SWZ_XOR(v, 1));
    v = fmaxf(v, SWZ_XOR(v, 2));
    v = fmaxf(v, SWZ_XOR(v, 4));
    v = fmaxf(v, SWZ_XOR(v, 8));
    return v;
}
__device__ __forceinline__ float redsum16(float v) {
    v += SWZ_XOR(v, 1);
    v += SWZ_XOR(v, 2);
    v += SWZ_XOR(v, 4);
    v += SWZ_XOR(v, 8);
    return v;
}

__device__ __forceinline__ v8f wmma_bf16(v16bf a, v16bf b, v8f c) {
    return __builtin_amdgcn_wmma_f32_16x16x32_bf16(false, a, false, b, (short)0, c, false, false);
}

// ---------------- conversion kernels ----------------
__global__ void k_cvt(const float* __restrict__ in, u16* __restrict__ out, int n) {
    int i = blockIdx.x * blockDim.x + threadIdx.x;
    if (i < n) out[i] = f2bf(in[i]);
}

// W [K][N] fp32 -> Wt [N][K] bf16  (so B-fragments read contiguous k)
__global__ void k_tcvt(const float* __restrict__ W, u16* __restrict__ Wt, int Kdim, int Ndim) {
    int i = blockIdx.x * blockDim.x + threadIdx.x;
    if (i < Kdim * Ndim) {
        int k = i / Ndim, n = i % Ndim;
        Wt[(size_t)n * Kdim + k] = f2bf(W[i]);
    }
}

// ---------------- bf16 WMMA GEMM ----------------
// A [M][K] bf16 row-major, Bt [N][K] bf16 (B transposed).
// Block = 4 waves; wave w computes a 64(row) x 32(col) tile:
//   rows [rowBase, rowBase+64), cols [blockIdx.x*128 + 32w, +32).
// Per 32-k step: 4 A-frags + 2 B-frags -> 8 WMMAs (~21 FLOP/B from cache).
// mode 1: scatter qkv into Q [B][H][T][D], K [B][H][T][D], Vt [B][H][D][T] (bf16)
// mode 2: write f32 out[M][N]
__global__ __launch_bounds__(128) void gemm_wmma(
    const u16* __restrict__ A, const u16* __restrict__ Bt,
    u16* __restrict__ qB, u16* __restrict__ kB, u16* __restrict__ vtB,
    float* __restrict__ outF, int M, int N, int K, int mode)
{
    const int lane = threadIdx.x & 31;
    const int wv   = threadIdx.x >> 5;
    const int ln   = lane & 15;
    const int lh   = lane >> 4;
    const int rowBase = blockIdx.y * 64;
    const int col     = blockIdx.x * 128 + wv * 32;

    v8f acc[4][2];
#pragma unroll
    for (int rt = 0; rt < 4; ++rt) {
        acc[rt][0] = vzero8();
        acc[rt][1] = vzero8();
    }

    const u16* bp0 = Bt + (size_t)(col + ln)      * K + lh * 16;
    const u16* bp1 = Bt + (size_t)(col + 16 + ln) * K + lh * 16;
    const u16* ap  = A  + (size_t)(rowBase + ln)  * K + lh * 16;

    for (int k = 0; k < K; k += 32) {
        v16bf bf0 = *(const v16bf*)(bp0 + k);
        v16bf bf1 = *(const v16bf*)(bp1 + k);
#pragma unroll
        for (int rt = 0; rt < 4; ++rt) {
            v16bf af = *(const v16bf*)(ap + (size_t)(rt * 16) * K + k);
            acc[rt][0] = wmma_bf16(af, bf0, acc[rt][0]);
            acc[rt][1] = wmma_bf16(af, bf1, acc[rt][1]);
        }
    }

#pragma unroll
    for (int rt = 0; rt < 4; ++rt) {
#pragma unroll
        for (int j = 0; j < 2; ++j) {
#pragma unroll
            for (int r = 0; r < 8; ++r) {
                int g = rowBase + rt * 16 + r + 8 * lh;   // global row (b*T + t)
                int n = col + j * 16 + ln;
                float val = acc[rt][j][r];
                if (mode == 1) {
                    int b = g >> 11;          // T = 2048
                    int t = g & 2047;
                    u16 bv = f2bf(val);
                    if (n < C_) {
                        int h = n >> 6, d = n & 63;
                        qB[(((size_t)b * H_ + h) * T_ + t) * D_ + d] = bv;
                    } else if (n < 2 * C_) {
                        int n2 = n - C_;
                        int h = n2 >> 6, d = n2 & 63;
                        kB[(((size_t)b * H_ + h) * T_ + t) * D_ + d] = bv;
                    } else {
                        int n3 = n - 2 * C_;
                        int h = n3 >> 6, d = n3 & 63;
                        vtB[(((size_t)b * H_ + h) * D_ + d) * T_ + t] = bv;  // V transposed
                    }
                } else {
                    outF[(size_t)g * N + n] = val;
                }
            }
        }
    }
}

// ---------------- flash attention ----------------
// Grid: B*H*(T/128) blocks of 128 threads (4 waves). Wave owns TWO 16-query
// tiles (32 queries) so each K/V fragment load feeds 2 WMMAs (~32 FLOP/B).
// Q,K: [B][H][T][D] bf16 ; Vt: [B][H][D][T] bf16 ; Y out: [B][T][C] bf16 (c=h*64+d)
__global__ __launch_bounds__(128) void flash_attn(
    const u16* __restrict__ Q, const u16* __restrict__ Kb,
    const u16* __restrict__ Vt, u16* __restrict__ Y)
{
    __shared__ __align__(64) u16 pTile[4][2][16 * 32];   // per-wave, per-qtile P

    const int lane = threadIdx.x & 31;
    const int wv   = threadIdx.x >> 5;
    const int ln   = lane & 15;
    const int lh   = lane >> 4;

    const int gid  = blockIdx.x;
    const int qBlk = gid & 15;          // T/128 = 16
    const int bh   = gid >> 4;
    const int b    = bh >> 4;           // H = 16
    const int hh   = bh & 15;
    const int qRow0 = qBlk * 128 + wv * 32;   // wave covers queries [qRow0, qRow0+32)

    const u16* qp = Q  + ((size_t)bh * T_ + qRow0) * D_;
    const u16* kp = Kb + (size_t)bh * T_ * D_;
    const u16* vp = Vt + (size_t)bh * D_ * T_;

    // Q A-fragments: [qtile][d-half]
    v16bf aq[2][2];
#pragma unroll
    for (int qt = 0; qt < 2; ++qt) {
        aq[qt][0] = *(const v16bf*)(qp + (qt * 16 + ln) * D_ + lh * 16);
        aq[qt][1] = *(const v16bf*)(qp + (qt * 16 + ln) * D_ + 32 + lh * 16);
    }

    v8f o[2][4];
    float mrow[2][8], lrow[2][8];
#pragma unroll
    for (int qt = 0; qt < 2; ++qt) {
#pragma unroll
        for (int dt = 0; dt < 4; ++dt) o[qt][dt] = vzero8();
#pragma unroll
        for (int r = 0; r < 8; ++r) { mrow[qt][r] = -1e30f; lrow[qt][r] = 0.f; }
    }

    // 32-key blocks covering keys <= qRow0+31
    const int nblk = (qRow0 + 63) >> 5;

    for (int blk = 0; blk < nblk; ++blk) {
        const int kb = blk * 32;
        const u16* kr = kp + (size_t)kb * D_;

        // K B-fragments, shared by both query tiles
        v16bf bk0a = *(const v16bf*)(kr + ln * D_ + lh * 16);
        v16bf bk0b = *(const v16bf*)(kr + ln * D_ + 32 + lh * 16);
        v16bf bk1a = *(const v16bf*)(kr + (16 + ln) * D_ + lh * 16);
        v16bf bk1b = *(const v16bf*)(kr + (16 + ln) * D_ + 32 + lh * 16);

#pragma unroll
        for (int qt = 0; qt < 2; ++qt) {
            v8f s0 = vzero8(), s1 = vzero8();
            s0 = wmma_bf16(aq[qt][0], bk0a, s0);
            s0 = wmma_bf16(aq[qt][1], bk0b, s0);
            s1 = wmma_bf16(aq[qt][0], bk1a, s1);
            s1 = wmma_bf16(aq[qt][1], bk1b, s1);

            const int qbase = qRow0 + qt * 16;
            float corr[8];
#pragma unroll
            for (int r = 0; r < 8; ++r) {
                int q  = qbase + r + 8 * lh;
                int k0 = kb + ln;
                float v0 = (k0      <= q) ? s0[r] * SC_ : -1e30f;
                float v1 = (k0 + 16 <= q) ? s1[r] * SC_ : -1e30f;
                float tmax = redmax16(fmaxf(v0, v1));
                float mnew = fmaxf(mrow[qt][r], tmax);
                corr[r] = exp2f(mrow[qt][r] - mnew);
                mrow[qt][r] = mnew;
                float p0 = exp2f(v0 - mnew);
                float p1 = exp2f(v1 - mnew);
                s0[r] = p0; s1[r] = p1;
                lrow[qt][r] = lrow[qt][r] * corr[r] + redsum16(p0 + p1);
            }
#pragma unroll
            for (int dt = 0; dt < 4; ++dt)
#pragma unroll
                for (int r = 0; r < 8; ++r) o[qt][dt][r] *= corr[r];

            // P (C-layout) -> LDS bf16 tile [16 rows][32 keys]
            u16* pl = pTile[wv][qt];
#pragma unroll
            for (int r = 0; r < 8; ++r) {
                pl[(r + 8 * lh) * 32 + ln]      = f2bf(s0[r]);
                pl[(r + 8 * lh) * 32 + 16 + ln] = f2bf(s1[r]);
            }
        }
        asm volatile("s_wait_dscnt 0x0" ::: "memory");

        // A-fragments of P: row = ln, k-half = lh*16
        v16bf pf0 = *(const v16bf*)(pTile[wv][0] + ln * 32 + lh * 16);
        v16bf pf1 = *(const v16bf*)(pTile[wv][1] + ln * 32 + lh * 16);
        asm volatile("" ::: "memory");

        // V B-fragments (shared by both query tiles), rows contiguous along keys
        const u16* vr = vp + kb + lh * 16;
#pragma unroll
        for (int dt = 0; dt < 4; ++dt) {
            v16bf vf = *(const v16bf*)(vr + (size_t)(dt * 16 + ln) * T_);
            o[0][dt] = wmma_bf16(pf0, vf, o[0][dt]);
            o[1][dt] = wmma_bf16(pf1, vf, o[1][dt]);
        }
    }

#pragma unroll
    for (int qt = 0; qt < 2; ++qt) {
        float inv[8];
#pragma unroll
        for (int r = 0; r < 8; ++r) inv[r] = 1.0f / lrow[qt][r];
#pragma unroll
        for (int dt = 0; dt < 4; ++dt) {
#pragma unroll
            for (int r = 0; r < 8; ++r) {
                int t = qRow0 + qt * 16 + r + 8 * lh;
                int c = hh * 64 + dt * 16 + ln;
                Y[((size_t)b * T_ + t) * C_ + c] = f2bf(o[qt][dt][r] * inv[r]);
            }
        }
    }
}

// ---------------- host side ----------------
extern "C" void kernel_launch(void* const* d_in, const int* in_sizes, int n_in,
                              void* d_out, int out_size, void* d_ws, size_t ws_size,
                              hipStream_t stream) {
    (void)in_sizes; (void)n_in; (void)out_size; (void)ws_size;
    const float* x      = (const float*)d_in[0];
    const float* W_attn = (const float*)d_in[1];
    const float* W_proj = (const float*)d_in[2];
    float* out = (float*)d_out;

    uint8_t* ws = (uint8_t*)d_ws;
    size_t off = 0;
    u16* xb   = (u16*)(ws + off); off += (size_t)B_ * T_ * C_ * 2;        // 16 MB
    u16* wat  = (u16*)(ws + off); off += (size_t)C3_ * C_ * 2;            // 6 MB
    u16* wpt  = (u16*)(ws + off); off += (size_t)C_ * C_ * 2;             // 2 MB
    u16* qb   = (u16*)(ws + off); off += (size_t)B_ * H_ * T_ * D_ * 2;   // 16 MB
    u16* kb   = (u16*)(ws + off); off += (size_t)B_ * H_ * T_ * D_ * 2;   // 16 MB
    u16* vtb  = (u16*)(ws + off); off += (size_t)B_ * H_ * D_ * T_ * 2;   // 16 MB
    u16* yb   = (u16*)(ws + off); off += (size_t)B_ * T_ * C_ * 2;        // 16 MB

    const int M = B_ * T_;  // 8192

    // 1) fp32 -> bf16 conversions (weights transposed)
    k_cvt <<<(M * C_ + 255) / 256, 256, 0, stream>>>(x, xb, M * C_);
    k_tcvt<<<(C_ * C3_ + 255) / 256, 256, 0, stream>>>(W_attn, wat, C_, C3_);
    k_tcvt<<<(C_ * C_  + 255) / 256, 256, 0, stream>>>(W_proj, wpt, C_, C_);

    // 2) qkv = x @ W_attn, scattered into Q/K/Vt (bf16)
    gemm_wmma<<<dim3(C3_ / 128, M / 64), 128, 0, stream>>>(
        xb, wat, qb, kb, vtb, nullptr, M, C3_, C_, 1);

    // 3) causal flash attention -> Y bf16 [B][T][C]
    flash_attn<<<B_ * H_ * (T_ / 128), 128, 0, stream>>>(qb, kb, vtb, yb);

    // 4) out = Y @ W_proj (f32)
    gemm_wmma<<<dim3(C_ / 128, M / 64), 128, 0, stream>>>(
        yb, wpt, nullptr, nullptr, nullptr, out, M, C_, C_, 2);
}
